// DGCNN_cls_89326729822452
// MI455X (gfx1250) — matrix operand order repaired and checked by reading, not compiled, not run on hardware
//
#include <hip/hip_runtime.h>
#include <math.h>

// ---------------------------------------------------------------------------
// DGCNN-cls forward for MI455X (gfx1250, wave32, WMMA).
// Heavy ops (edge-conv 1x1, fused-7x7 SKN conv, conv5) are implicit GEMMs on
// v_wmma_f32_16x16x32_f16 with 4x M-tile register blocking (one gathered B
// fragment feeds 4 WMMAs), double-buffered LDS-staged A operands (one barrier
// per K-chunk, staging overlapped with WMMA), and global_prefetch of the K+2
// weight chunk. SKN's 3 branches are fused into one effective 7x7 filter.
// ---------------------------------------------------------------------------

typedef __attribute__((ext_vector_type(16))) _Float16 v16h;
typedef __attribute__((ext_vector_type(8)))  _Float16 v8h;
typedef __attribute__((ext_vector_type(8)))  float    v8f;

union AFrag { v16h v; v8h h[2]; };

#define KNN 20
#define EPSV 1e-5f

// Stage one 64x32-half A chunk into LDS (cooperative, 8 halves per thread).
// Expects locals: wrow, ldr, ldc, vecok, KTOT in scope.
#define STAGE_A(dstbuf, kcv)                                                \
  do {                                                                      \
    if (vecok && (kcv) + 32 <= KTOT) {                                      \
      *(v8h*)&(dstbuf)[ldr * 32 + ldc] = *(const v8h*)&wrow[(kcv) + ldc];   \
    } else {                                                                \
      for (int j_ = 0; j_ < 8; ++j_) {                                      \
        int k_ = (kcv) + ldc + j_;                                          \
        (dstbuf)[ldr * 32 + ldc + j_] =                                     \
            (k_ < KTOT) ? wrow[k_] : (_Float16)0.f;                         \
      }                                                                     \
    }                                                                       \
  } while (0)

// ------------------------------- small utils -------------------------------
__global__ void cvt_f16_kernel(const float* __restrict__ src,
                               _Float16* __restrict__ dst, long n) {
  long i = (long)blockIdx.x * blockDim.x + threadIdx.x;
  if (i < n) dst[i] = (_Float16)src[i];
}

// Fuse 3x3/5x5/7x7 branches into one 7x7 filter (f16) + fused bias.
__global__ void skn_fuse_kernel(const float* __restrict__ w3,
                                const float* __restrict__ w5,
                                const float* __restrict__ w7,
                                const float* __restrict__ b3,
                                const float* __restrict__ b5,
                                const float* __restrict__ b7,
                                const float* __restrict__ alpha,
                                _Float16* __restrict__ weff,
                                float* __restrict__ beff, int O, int C) {
  int i = blockIdx.x * blockDim.x + threadIdx.x;
  int total = O * C * 49;
  if (i >= total) return;
  int r = i % 49, oc = i / 49;
  int c = oc % C, o = oc / C;
  int dy = r / 7 - 3, dx = r % 7 - 3;
  float a0 = alpha[0], a1 = alpha[1], a2 = alpha[2];
  float v = a2 * w7[((size_t)(o * C + c) * 7 + (dy + 3)) * 7 + (dx + 3)];
  if (dy >= -2 && dy <= 2 && dx >= -2 && dx <= 2)
    v += a1 * w5[((size_t)(o * C + c) * 5 + (dy + 2)) * 5 + (dx + 2)];
  if (dy >= -1 && dy <= 1 && dx >= -1 && dx <= 1)
    v += a0 * w3[((size_t)(o * C + c) * 3 + (dy + 1)) * 3 + (dx + 1)];
  weff[(size_t)(o * C + c) * 49 + r] = (_Float16)v;
  if (i < O) beff[i] = a0 * b3[i] + a1 * b5[i] + a2 * b7[i];
}

// --------------------------------- kNN -------------------------------------
// one wave per (b,n): per-lane top-20 insertion, then 20x wave-wide argmax.
__global__ __launch_bounds__(256) void knn_kernel(
    const float* __restrict__ xin, int Ctot, int coff, int C, int N, int B,
    int* __restrict__ idxout) {
  __shared__ float ctr[8][128];
  int lane = threadIdx.x & 31, wv = threadIdx.x >> 5;
  int row = blockIdx.x * 8 + wv;
  int b = row / N, n = row % N;
  const float* xb = xin + ((size_t)b * Ctot + coff) * N;
  for (int c = lane; c < C; c += 32) ctr[wv][c] = xb[(size_t)c * N + n];
  __syncthreads();
  float xxn = 0.f;
  for (int c = 0; c < C; ++c) xxn += ctr[wv][c] * ctr[wv][c];
  float best[KNN]; int bidx[KNN];
  for (int j = 0; j < KNN; ++j) { best[j] = -3.4e38f; bidx[j] = 0; }
  for (int m = lane; m < N; m += 32) {
    float inner = 0.f, xxm = 0.f;
    for (int c = 0; c < C; ++c) {
      float xm = xb[(size_t)c * N + m];
      inner += xm * ctr[wv][c];
      xxm += xm * xm;
    }
    float d = 2.f * inner - xxn - xxm;
    if (d > best[KNN - 1]) {
      int j = KNN - 1;
      while (j > 0 && best[j - 1] < d) {
        best[j] = best[j - 1]; bidx[j] = bidx[j - 1]; --j;
      }
      best[j] = d; bidx[j] = m;
    }
  }
  int ptr = 0;
  for (int j = 0; j < KNN; ++j) {
    float v = (ptr < KNN) ? best[ptr] : -3.4e38f;
    int vi = (ptr < KNN) ? bidx[ptr] : 0;
    int vl = lane;
    for (int off = 16; off > 0; off >>= 1) {
      float ov = __shfl_xor(v, off, 32);
      int oi = __shfl_xor(vi, off, 32);
      int ol = __shfl_xor(vl, off, 32);
      if (ov > v || (ov == v && ol < vl)) { v = ov; vi = oi; vl = ol; }
    }
    if (lane == vl) ptr++;
    if (lane == 0) idxout[(size_t)row * KNN + j] = vi;
  }
}

// ------------------------- EdgeConv implicit GEMM ---------------------------
// y[b,o,n,kk] = sum_c W[o,c] * feat[b,n,kk,c],  feat = [nb-ctr | ctr], K2=2Cin
// Wave computes 64 M x 16 pixels (4 WMMA tiles sharing one B fragment).
__global__ __launch_bounds__(256) void edgeconv_wmma_kernel(
    const _Float16* __restrict__ xh, int Ctot, int coff,
    const int* __restrict__ idx, const _Float16* __restrict__ w,
    float* __restrict__ y, int B, int Cin, int Cout, int N) {
  const int KTOT = 2 * Cin;
  __shared__ _Float16 atile[2][64 * 32];
  const int tid = threadIdx.x;
  const int lane = tid & 31, wv = tid >> 5;
  const int m0 = blockIdx.y * 64;
  const int p0 = (blockIdx.x * 8 + wv) * 16;
  int p = p0 + (lane & 15);
  int b = p / (N * KNN);
  int rem = p - b * (N * KNN);
  int n = rem / KNN, kk = rem - n * KNN;
  int nbr = idx[((size_t)b * N + n) * KNN + kk];
  const _Float16* xb = xh + ((size_t)b * Ctot + coff) * N;
  const int koff = (lane & 16) ? 16 : 0;
  const int abase = (lane & 16) ? 16 : 0;
  const int arow = lane & 15;
  const int ldr = tid >> 2;        // 0..63 (tile row)
  const int ldc = (tid & 3) * 8;   // 0,8,16,24
  const bool vecok = (KTOT % 8) == 0;
  const _Float16* wrow = w + (size_t)(m0 + ldr) * KTOT;
  v8f acc[4] = {};
  STAGE_A(atile[0], 0);
  __syncthreads();
  int buf = 0;
  for (int kc = 0; kc < KTOT; kc += 32) {
    if (kc + 32 < KTOT) STAGE_A(atile[buf ^ 1], kc + 32);  // overlap w/ WMMA
    if (kc + 64 < KTOT) __builtin_prefetch(&wrow[kc + 64 + ldc], 0, 3);
    // gathered B fragment (shared by 4 M-tiles)
    v16h bf;
    int kbase = kc + koff;
    if (kbase + 16 <= Cin) {
#pragma unroll
      for (int e = 0; e < 16; ++e) {
        int k = kbase + e;
        bf[e] = (_Float16)(xb[(size_t)k * N + nbr] - xb[(size_t)k * N + n]);
      }
    } else if (kbase >= Cin) {
#pragma unroll
      for (int e = 0; e < 16; ++e) {
        int k = kbase + e;
        _Float16 v = (_Float16)0.f;
        if (k < KTOT) v = xb[(size_t)(k - Cin) * N + n];
        bf[e] = v;
      }
    } else {
#pragma unroll
      for (int e = 0; e < 16; ++e) {
        int k = kbase + e;
        _Float16 v = (_Float16)0.f;
        if (k < Cin)
          v = (_Float16)(xb[(size_t)k * N + nbr] - xb[(size_t)k * N + n]);
        else if (k < KTOT)
          v = xb[(size_t)(k - Cin) * N + n];
        bf[e] = v;
      }
    }
    AFrag af[4];
#pragma unroll
    for (int t = 0; t < 4; ++t) {
      af[t].h[0] = *(const v8h*)&atile[buf][(arow + 16 * t) * 32 + abase];
      af[t].h[1] = *(const v8h*)&atile[buf][(arow + 16 * t) * 32 + 8 + abase];
    }
#pragma unroll
    for (int t = 0; t < 4; ++t)
      acc[t] = __builtin_amdgcn_wmma_f32_16x16x32_f16(
          false, af[t].v, false, bf, (short)0, acc[t], false, false);
    buf ^= 1;
    __syncthreads();
  }
#pragma unroll
  for (int t = 0; t < 4; ++t)
#pragma unroll
    for (int r = 0; r < 8; ++r) {
      int m = m0 + 16 * t + r + ((lane & 16) ? 8 : 0);
      y[(((size_t)b * Cout + m) * N + n) * KNN + kk] = acc[t][r];
    }
}

// --------------------- fused-7x7 SKN conv implicit GEMM ---------------------
// out[b,o,n,kk] = sum_{c,dy,dx} Weff[o, c*49+dy*7+dx] * xh[b,c,n+dy-3,kk+dx-3]
__global__ __launch_bounds__(256) void skn_wmma_kernel(
    const _Float16* __restrict__ xh, const _Float16* __restrict__ weff,
    const float* __restrict__ beff, float* __restrict__ out, int B, int C,
    int N) {
  const int KTOT = C * 49;  // 3136 or 6272: multiple of 32
  __shared__ _Float16 atile[2][64 * 32];
  const int tid = threadIdx.x;
  const int lane = tid & 31, wv = tid >> 5;
  const int m0 = blockIdx.y * 64;
  const int p0 = (blockIdx.x * 8 + wv) * 16;
  int p = p0 + (lane & 15);
  int b = p / (N * KNN);
  int rem = p - b * (N * KNN);
  int n = rem / KNN, kk = rem - n * KNN;
  const int koff = (lane & 16) ? 16 : 0;
  const int abase = (lane & 16) ? 16 : 0;
  const int arow = lane & 15;
  const int ldr = tid >> 2;
  const int ldc = (tid & 3) * 8;
  const bool vecok = true;
  const _Float16* wrow = weff + (size_t)(m0 + ldr) * KTOT;
  v8f acc[4] = {};
  STAGE_A(atile[0], 0);
  __syncthreads();
  int buf = 0;
  for (int kc = 0; kc < KTOT; kc += 32) {
    if (kc + 32 < KTOT) STAGE_A(atile[buf ^ 1], kc + 32);
    if (kc + 64 < KTOT) __builtin_prefetch(&wrow[kc + 64 + ldc], 0, 3);
    v16h bf;
    int q0 = kc + koff;
    int c0 = q0 / 49;          // constant-divisor mul-shift, once per chunk
    int r0 = q0 - c0 * 49;
#pragma unroll
    for (int e = 0; e < 16; ++e) {
      int r = r0 + e, c = c0;
      if (r >= 49) { r -= 49; ++c; }      // r0+15 < 64 < 98: one step enough
      int dy = r / 7;
      int dx = r - dy * 7;
      int nn = n + dy - 3, kx = kk + dx - 3;
      _Float16 v = (_Float16)0.f;
      if (nn >= 0 && nn < N && kx >= 0 && kx < KNN)
        v = xh[(((size_t)b * C + c) * N + nn) * KNN + kx];
      bf[e] = v;
    }
    AFrag af[4];
#pragma unroll
    for (int t = 0; t < 4; ++t) {
      af[t].h[0] = *(const v8h*)&atile[buf][(arow + 16 * t) * 32 + abase];
      af[t].h[1] = *(const v8h*)&atile[buf][(arow + 16 * t) * 32 + 8 + abase];
    }
#pragma unroll
    for (int t = 0; t < 4; ++t)
      acc[t] = __builtin_amdgcn_wmma_f32_16x16x32_f16(
          false, af[t].v, false, bf, (short)0, acc[t], false, false);
    buf ^= 1;
    __syncthreads();
  }
#pragma unroll
  for (int t = 0; t < 4; ++t)
#pragma unroll
    for (int r = 0; r < 8; ++r) {
      int m = m0 + 16 * t + r + ((lane & 16) ? 8 : 0);
      out[(((size_t)b * C + m) * N + n) * KNN + kk] = acc[t][r] + beff[m];
    }
}

// ------------------------------ conv5 GEMM ----------------------------------
// e[b,o,n] = sum_c cat[b,c,n] * W[o,c];  M=1024, K=512, pixels = B*N
__global__ __launch_bounds__(256) void conv5_wmma_kernel(
    const _Float16* __restrict__ cath, const _Float16* __restrict__ w,
    float* __restrict__ e5, int B, int N) {
  const int KTOT = 512;
  __shared__ _Float16 atile[2][64 * 32];
  const int tid = threadIdx.x;
  const int lane = tid & 31, wv = tid >> 5;
  const int m0 = blockIdx.y * 64;
  const int p0 = (blockIdx.x * 8 + wv) * 16;
  int p = p0 + (lane & 15);
  int b = p / N, n = p - b * N;
  const int koff = (lane & 16) ? 16 : 0;
  const int abase = (lane & 16) ? 16 : 0;
  const int arow = lane & 15;
  const int ldr = tid >> 2;
  const int ldc = (tid & 3) * 8;
  const bool vecok = true;
  const _Float16* wrow = w + (size_t)(m0 + ldr) * KTOT;
  v8f acc[4] = {};
  STAGE_A(atile[0], 0);
  __syncthreads();
  int buf = 0;
  for (int kc = 0; kc < KTOT; kc += 32) {
    if (kc + 32 < KTOT) STAGE_A(atile[buf ^ 1], kc + 32);
    if (kc + 64 < KTOT) __builtin_prefetch(&wrow[kc + 64 + ldc], 0, 3);
    v16h bf;
#pragma unroll
    for (int e = 0; e < 16; ++e)
      bf[e] = cath[((size_t)b * 512 + kc + koff + e) * N + n];
    AFrag af[4];
#pragma unroll
    for (int t = 0; t < 4; ++t) {
      af[t].h[0] = *(const v8h*)&atile[buf][(arow + 16 * t) * 32 + abase];
      af[t].h[1] = *(const v8h*)&atile[buf][(arow + 16 * t) * 32 + 8 + abase];
    }
#pragma unroll
    for (int t = 0; t < 4; ++t)
      acc[t] = __builtin_amdgcn_wmma_f32_16x16x32_f16(
          false, af[t].v, false, bf, (short)0, acc[t], false, false);
    buf ^= 1;
    __syncthreads();
  }
#pragma unroll
  for (int t = 0; t < 4; ++t)
#pragma unroll
    for (int r = 0; r < 8; ++r) {
      int m = m0 + 16 * t + r + ((lane & 16) ? 8 : 0);
      e5[((size_t)b * 1024 + m) * N + n] = acc[t][r];
    }
}

// --------------------------- BN (training mode) -----------------------------
__global__ void bn_stats_kernel(const float* __restrict__ y,
                                float* __restrict__ stats, int C, int S,
                                int B) {
  int o = blockIdx.x, tid = threadIdx.x;
  float s0 = 0.f, s1 = 0.f;
  for (long t = tid; t < (long)B * S; t += 256) {
    int b = (int)(t / S);
    long sp = t - (long)b * S;
    float v = y[((size_t)b * C + o) * S + sp];
    s0 += v; s1 += v * v;
  }
  __shared__ float r0[256], r1[256];
  r0[tid] = s0; r1[tid] = s1;
  __syncthreads();
  for (int st = 128; st > 0; st >>= 1) {
    if (tid < st) { r0[tid] += r0[tid + st]; r1[tid] += r1[tid + st]; }
    __syncthreads();
  }
  if (tid == 0) {
    float cnt = (float)B * (float)S;
    float m = r0[0] / cnt;
    float var = r1[0] / cnt - m * m;
    stats[o] = m;
    stats[C + o] = rsqrtf(var + EPSV);
  }
}

__global__ void bn_apply_leaky_kernel(float* __restrict__ y,
                                      const float* __restrict__ stats,
                                      const float* __restrict__ g,
                                      const float* __restrict__ bb, int C,
                                      int S, long total) {
  long i = (long)blockIdx.x * blockDim.x + threadIdx.x;
  if (i >= total) return;
  int o = (int)((i / S) % C);
  float v = g[o] * (y[i] - stats[o]) * stats[C + o] + bb[o];
  y[i] = v > 0.f ? v : 0.2f * v;
}

// --------------------------------- SE ---------------------------------------
__global__ void se_pool_kernel(const float* __restrict__ y,
                               float* __restrict__ z, int C, int S) {
  int bc = blockIdx.x, tid = threadIdx.x;
  const float* p = y + (size_t)bc * S;
  float acc = 0.f;
  for (int i = tid; i < S; i += 256) acc += p[i];
  __shared__ float r[256];
  r[tid] = acc;
  __syncthreads();
  for (int st = 128; st > 0; st >>= 1) {
    if (tid < st) r[tid] += r[tid + st];
    __syncthreads();
  }
  if (tid == 0) z[bc] = r[0] / (float)S;
}

__global__ void se_fc_kernel(const float* __restrict__ z,
                             const float* __restrict__ w1,
                             const float* __restrict__ w2,
                             float* __restrict__ s, int C, int R) {
  int b = blockIdx.x, tid = threadIdx.x;
  __shared__ float h[32];
  if (tid < R) {
    float a = 0.f;
    for (int c = 0; c < C; ++c) a += w1[(size_t)tid * C + c] * z[(size_t)b * C + c];
    h[tid] = a > 0.f ? a : 0.f;
  }
  __syncthreads();
  if (tid < C) {
    float a = 0.f;
    for (int r = 0; r < R; ++r) a += w2[(size_t)tid * R + r] * h[r];
    s[(size_t)b * C + tid] = 1.f / (1.f + expf(-a));
  }
}

// scale by SE gate and convert to f16 (SKN conv input)
__global__ void scale_half_kernel(const float* __restrict__ y,
                                  const float* __restrict__ s,
                                  _Float16* __restrict__ dst, int C, int S,
                                  long total) {
  long i = (long)blockIdx.x * blockDim.x + threadIdx.x;
  if (i >= total) return;
  int c = (int)((i / S) % C);
  int b = (int)(i / ((long)S * C));
  dst[i] = (_Float16)(y[i] * s[(size_t)b * C + c]);
}

// max over k axis; optional SE gate (sigmoid>0 commutes with max).
// Dual-writes the f32 concat buffer (for kNN) and its f16 mirror (for WMMA).
__global__ void maxk_kernel(const float* __restrict__ src,
                            const float* __restrict__ s,
                            float* __restrict__ cat,
                            _Float16* __restrict__ cath, int C, int coff,
                            int N, int B) {
  int t = blockIdx.x * blockDim.x + threadIdx.x;
  if (t >= B * C * N) return;
  int n = t % N;
  int c = (t / N) % C;
  int b = t / (N * C);
  const float* p = src + (((size_t)b * C + c) * N + n) * KNN;
  float m = -3.4e38f;
  for (int k = 0; k < KNN; ++k) m = fmaxf(m, p[k]);
  if (s) m *= s[(size_t)b * C + c];
  size_t o = ((size_t)b * 512 + coff + c) * N + n;
  cat[o] = m;
  cath[o] = (_Float16)m;
}

// global max + mean pool of e5 -> g = [gmax | gavg]
__global__ void gpool_kernel(const float* __restrict__ e5,
                             float* __restrict__ g, int Cemb, int N) {
  int bo = blockIdx.x, tid = threadIdx.x;
  int b = bo / Cemb, o = bo % Cemb;
  const float* p = e5 + ((size_t)b * Cemb + o) * N;
  float mx = -3.4e38f, sm = 0.f;
  for (int i = tid; i < N; i += 256) { mx = fmaxf(mx, p[i]); sm += p[i]; }
  __shared__ float rm[256], rs[256];
  rm[tid] = mx; rs[tid] = sm;
  __syncthreads();
  for (int st = 128; st > 0; st >>= 1) {
    if (tid < st) { rm[tid] = fmaxf(rm[tid], rm[tid + st]); rs[tid] += rs[tid + st]; }
    __syncthreads();
  }
  if (tid == 0) {
    g[(size_t)b * 2 * Cemb + o] = rm[0];
    g[(size_t)b * 2 * Cemb + Cemb + o] = rs[0] / (float)N;
  }
}

// ----------------------------- tail MLP -------------------------------------
__global__ void lin_kernel(const float* __restrict__ in,
                           const float* __restrict__ W,
                           const float* __restrict__ bias,
                           float* __restrict__ out, int B, int J, int K) {
  int t = blockIdx.x * blockDim.x + threadIdx.x;
  if (t >= B * J) return;
  int b = t / J, j = t % J;
  float acc = bias ? bias[j] : 0.f;
  for (int k = 0; k < K; ++k) acc += in[(size_t)b * K + k] * W[(size_t)j * K + k];
  out[t] = acc;
}

__global__ void bn_batch_leaky_kernel(float* __restrict__ x,
                                      const float* __restrict__ g,
                                      const float* __restrict__ bb, int B,
                                      int J) {
  int j = blockIdx.x * blockDim.x + threadIdx.x;
  if (j >= J) return;
  float m = 0.f;
  for (int b = 0; b < B; ++b) m += x[(size_t)b * J + j];
  m /= (float)B;
  float v = 0.f;
  for (int b = 0; b < B; ++b) {
    float d = x[(size_t)b * J + j] - m;
    v += d * d;
  }
  v /= (float)B;
  float is = rsqrtf(v + EPSV);
  for (int b = 0; b < B; ++b) {
    float y = g[j] * (x[(size_t)b * J + j] - m) * is + bb[j];
    x[(size_t)b * J + j] = y > 0.f ? y : 0.2f * y;
  }
}

// ---------------------------------------------------------------------------
extern "C" void kernel_launch(void* const* d_in, const int* in_sizes, int n_in,
                              void* d_out, int out_size, void* d_ws,
                              size_t ws_size, hipStream_t stream) {
  (void)in_sizes; (void)n_in; (void)out_size; (void)ws_size;
  const int B = 4, N = 1024;
  const float* x0 = (const float*)d_in[0];
  const float* conv_w[4] = {(const float*)d_in[1], (const float*)d_in[4],
                            (const float*)d_in[7], (const float*)d_in[10]};
  const float* bn_g[4] = {(const float*)d_in[2], (const float*)d_in[5],
                          (const float*)d_in[8], (const float*)d_in[11]};
  const float* bn_b[4] = {(const float*)d_in[3], (const float*)d_in[6],
                          (const float*)d_in[9], (const float*)d_in[12]};
  const float* se_w1[4] = {(const float*)d_in[13], (const float*)d_in[15],
                           (const float*)d_in[17], (const float*)d_in[19]};
  const float* se_w2[4] = {(const float*)d_in[14], (const float*)d_in[16],
                           (const float*)d_in[18], (const float*)d_in[20]};
  const float* skn_w[3][3] = {
      {(const float*)d_in[21], (const float*)d_in[22], (const float*)d_in[23]},
      {(const float*)d_in[28], (const float*)d_in[29], (const float*)d_in[30]},
      {(const float*)d_in[35], (const float*)d_in[36], (const float*)d_in[37]}};
  const float* skn_b[3][3] = {
      {(const float*)d_in[24], (const float*)d_in[25], (const float*)d_in[26]},
      {(const float*)d_in[31], (const float*)d_in[32], (const float*)d_in[33]},
      {(const float*)d_in[38], (const float*)d_in[39], (const float*)d_in[40]}};
  const float* skn_a[3] = {(const float*)d_in[27], (const float*)d_in[34],
                           (const float*)d_in[41]};
  const float* conv5_w = (const float*)d_in[42];
  const float* bn5_g = (const float*)d_in[43];
  const float* bn5_b = (const float*)d_in[44];
  const float* lin1_w = (const float*)d_in[45];
  const float* bn6_g = (const float*)d_in[46];
  const float* bn6_b = (const float*)d_in[47];
  const float* lin2_w = (const float*)d_in[48];
  const float* lin2_b = (const float*)d_in[49];
  const float* bn7_g = (const float*)d_in[50];
  const float* bn7_b = (const float*)d_in[51];
  const float* lin3_w = (const float*)d_in[52];
  const float* lin3_b = (const float*)d_in[53];

  // ---- workspace carving ----
  char* ws = (char*)d_ws;
  size_t off = 0;
  auto alloc = [&](size_t bytes) -> void* {
    off = (off + 255) & ~(size_t)255;
    void* p = ws + off;
    off += bytes;
    return p;
  };
  int* idx = (int*)alloc((size_t)B * N * KNN * 4);
  float* y = (float*)alloc((size_t)B * 256 * N * KNN * 4);
  float* y2 = (float*)alloc((size_t)B * 128 * N * KNN * 4);
  _Float16* yh = (_Float16*)alloc((size_t)B * 128 * N * KNN * 2);
  float* stats = (float*)alloc(2 * 1024 * 4);
  float* z = (float*)alloc((size_t)B * 256 * 4);
  float* sgate = (float*)alloc((size_t)B * 256 * 4);
  float* cat = (float*)alloc((size_t)B * 512 * N * 4);
  _Float16* cath = (_Float16*)alloc((size_t)B * 512 * N * 2);
  _Float16* x0h = (_Float16*)alloc((size_t)B * 3 * N * 2);
  float* e5 = (float*)alloc((size_t)B * 1024 * N * 4);
  _Float16* wh[4];
  const int wsz[4] = {64 * 6, 64 * 128, 128 * 128, 256 * 256};
  for (int i = 0; i < 4; ++i) wh[i] = (_Float16*)alloc((size_t)wsz[i] * 2);
  _Float16* w5h = (_Float16*)alloc((size_t)1024 * 512 * 2);
  _Float16* weff[3];
  float* beff[3];
  const int sknC[3] = {64, 64, 128};
  for (int i = 0; i < 3; ++i) {
    weff[i] = (_Float16*)alloc((size_t)sknC[i] * sknC[i] * 49 * 2);
    beff[i] = (float*)alloc((size_t)sknC[i] * 4);
  }
  float* g = (float*)alloc((size_t)B * 2048 * 4);
  float* h1 = (float*)alloc((size_t)B * 512 * 4);
  float* h2 = (float*)alloc((size_t)B * 256 * 4);

  // ---- prep: f16 weights, f16 stage-1 input, SKN branch fusion ----
  for (int i = 0; i < 4; ++i)
    cvt_f16_kernel<<<(wsz[i] + 255) / 256, 256, 0, stream>>>(conv_w[i], wh[i], wsz[i]);
  cvt_f16_kernel<<<(1024 * 512 + 255) / 256, 256, 0, stream>>>(conv5_w, w5h, 1024 * 512);
  cvt_f16_kernel<<<(B * 3 * N + 255) / 256, 256, 0, stream>>>(x0, x0h, (long)B * 3 * N);
  for (int i = 0; i < 3; ++i) {
    int C = sknC[i], tot = C * C * 49;
    skn_fuse_kernel<<<(tot + 255) / 256, 256, 0, stream>>>(
        skn_w[i][0], skn_w[i][1], skn_w[i][2], skn_b[i][0], skn_b[i][1],
        skn_b[i][2], skn_a[i], weff[i], beff[i], C, C);
  }

  // ---- 4 edge-conv stages ----
  const int S = N * KNN;            // 20480
  const int P = B * S;              // 81920
  const int stCin[4] = {3, 64, 64, 128};
  const int stCout[4] = {64, 64, 128, 256};
  const int stCoffIn[4] = {0, 0, 64, 128};
  const int stCoffOut[4] = {0, 64, 128, 256};
  const int seR[4] = {4, 4, 8, 16};

  for (int st = 0; st < 4; ++st) {
    const float* xin_f32 = (st == 0) ? x0 : cat;      // kNN distance input
    const _Float16* xin_h = (st == 0) ? x0h : cath;   // GEMM gather input
    int Ctot = (st == 0) ? 3 : 512;
    int coff = (st == 0) ? 0 : stCoffIn[st];
    int Cin = stCin[st], Cout = stCout[st];
    long tot = (long)B * Cout * S;

    knn_kernel<<<(B * N) / 8, 256, 0, stream>>>(xin_f32, Ctot, coff, Cin, N, B, idx);
    edgeconv_wmma_kernel<<<dim3(P / 128, Cout / 64), 256, 0, stream>>>(
        xin_h, Ctot, coff, idx, wh[st], y, B, Cin, Cout, N);
    bn_stats_kernel<<<Cout, 256, 0, stream>>>(y, stats, Cout, S, B);
    bn_apply_leaky_kernel<<<(int)((tot + 255) / 256), 256, 0, stream>>>(
        y, stats, bn_g[st], bn_b[st], Cout, S, tot);
    se_pool_kernel<<<B * Cout, 256, 0, stream>>>(y, z, Cout, S);
    se_fc_kernel<<<B, 256, 0, stream>>>(z, se_w1[st], se_w2[st], sgate, Cout, seR[st]);

    if (st < 3) {
      scale_half_kernel<<<(int)((tot + 255) / 256), 256, 0, stream>>>(
          y, sgate, yh, Cout, S, tot);
      skn_wmma_kernel<<<dim3(P / 128, Cout / 64), 256, 0, stream>>>(
          yh, weff[st], beff[st], y2, B, Cout, N);
      maxk_kernel<<<(B * Cout * N + 255) / 256, 256, 0, stream>>>(
          y2, nullptr, cat, cath, Cout, stCoffOut[st], N, B);
    } else {
      // stage 4: SE gate folded into the k-max (gate > 0 commutes with max)
      maxk_kernel<<<(B * Cout * N + 255) / 256, 256, 0, stream>>>(
          y, sgate, cat, cath, Cout, stCoffOut[st], N, B);
    }
  }

  // ---- conv5 (512 -> 1024) + BN/leaky + global pooling ----
  conv5_wmma_kernel<<<dim3((B * N) / 128, 1024 / 64), 256, 0, stream>>>(
      cath, w5h, e5, B, N);
  bn_stats_kernel<<<1024, 256, 0, stream>>>(e5, stats, 1024, N, B);
  bn_apply_leaky_kernel<<<(int)(((long)B * 1024 * N + 255) / 256), 256, 0, stream>>>(
      e5, stats, bn5_g, bn5_b, 1024, N, (long)B * 1024 * N);
  gpool_kernel<<<B * 1024, 256, 0, stream>>>(e5, g, 1024, N);

  // ---- tail MLP ----
  lin_kernel<<<(B * 512 + 255) / 256, 256, 0, stream>>>(g, lin1_w, nullptr, h1, B, 512, 2048);
  bn_batch_leaky_kernel<<<2, 256, 0, stream>>>(h1, bn6_g, bn6_b, B, 512);
  lin_kernel<<<(B * 256 + 255) / 256, 256, 0, stream>>>(h1, lin2_w, lin2_b, h2, B, 256, 512);
  bn_batch_leaky_kernel<<<1, 256, 0, stream>>>(h2, bn7_g, bn7_b, B, 256);
  lin_kernel<<<1, 256, 0, stream>>>(h2, lin3_w, lin3_b, (float*)d_out, B, 40, 256);
}
